// GCN_73933567033777
// MI455X (gfx1250) — matrix operand order repaired and checked by reading, not compiled, not run on hardware
//
#include <hip/hip_runtime.h>
#include <hip/hip_bf16.h>

typedef __attribute__((ext_vector_type(16))) __bf16 v16bf;
typedef __attribute__((ext_vector_type(8)))  float  v8f;

// ---------- degree / norm ----------
__global__ void init_deg_kernel(float* __restrict__ deg, int N) {
  int i = blockIdx.x * blockDim.x + threadIdx.x;
  if (i < N) deg[i] = 1.0f;  // self-loop weight = 1
}

__global__ void deg_accum_kernel(const int* __restrict__ dst, const float* __restrict__ ew,
                                 float* __restrict__ deg, int E) {
  int e = blockIdx.x * blockDim.x + threadIdx.x;
  if (e < E) unsafeAtomicAdd(&deg[dst[e]], ew[e]);
}

__global__ void dinv_kernel(float* __restrict__ deg, int N) {
  int i = blockIdx.x * blockDim.x + threadIdx.x;
  if (i < N) {
    float d = deg[i];
    deg[i] = d > 0.0f ? rsqrtf(d) : 0.0f;   // becomes dinv in place
  }
}

// ---------- dense GEMM via bf16 WMMA, f32 accumulate ----------
// C[M, NT*16] = A[M,K] * B[K, NT*16], f32 row-major in memory, bf16 WMMA.
// One wave per 16-row strip of C; the A fragment is loaded once per K-step and
// reused across all NT column tiles (NT accumulators live in registers).
// Requires M%16==0, K%32==0.
template <int NT>
__global__ void gemm_bf16_wmma_kernel(const float* __restrict__ A, const float* __restrict__ B,
                                      float* __restrict__ C, int M, int K) {
  const int N = NT * 16;
  const int mtiles = M >> 4;
  int wave = (int)(((long long)blockIdx.x * blockDim.x + threadIdx.x) >> 5);
  if (wave >= mtiles) return;                 // wave-uniform exit: EXEC all-1 for WMMA
  const int lane = threadIdx.x & 31;
  const int half = lane >> 4;                 // 0: lanes 0-15, 1: lanes 16-31
  const int r    = lane & 15;
  const float* arow = A + (size_t)(wave * 16 + r) * K;   // A row for this lane
  v8f acc[NT];
#pragma unroll
  for (int t = 0; t < NT; ++t) acc[t] = (v8f){};

  for (int k0 = 0; k0 < K; k0 += 32) {
    v16bf a;
    const int ak = k0 + half * 16;   // contiguous K: 16 K per half, matching B
#pragma unroll
    for (int i = 0; i < 8; ++i) {
      a[i]     = (__bf16)arow[ak + i];
      a[i + 8] = (__bf16)arow[ak + 8 + i];
    }
    const int bk = k0 + half * 16;   // contiguous K: 16 K rows per half, matching A
#pragma unroll
    for (int t = 0; t < NT; ++t) {
      v16bf b;
      const int bcol = t * 16 + r;
#pragma unroll
      for (int i = 0; i < 16; ++i)
        b[i] = (__bf16)B[(size_t)(bk + i) * N + bcol];
      acc[t] = __builtin_amdgcn_wmma_f32_16x16x32_bf16(
          /*neg_a=*/false, a, /*neg_b=*/false, b,
          /*c_mod=*/(short)0, acc[t], /*reuse_a=*/false, /*reuse_b=*/false);
    }
  }
  // C layout: lane half selects M rows 0-7 / 8-15; lane%16 = N column
#pragma unroll
  for (int t = 0; t < NT; ++t) {
    float* crow = C + (size_t)(wave * 16 + half * 8) * N + (size_t)t * 16 + r;
#pragma unroll
    for (int v = 0; v < 8; ++v) crow[(size_t)v * N] = acc[t][v];
  }
}

// ---------- self-loop contribution (initializes accumulator, no atomics needed) ----------
template <int F>
__global__ void selfloop_init_kernel(const float* __restrict__ xw, const float* __restrict__ dinv,
                                     const float* __restrict__ bias, float* __restrict__ out,
                                     int N, int addBias) {
  long long i = (long long)blockIdx.x * blockDim.x + threadIdx.x;
  if (i >= (long long)N * F) return;
  int node = (int)(i / F);
  int f    = (int)(i % F);
  float di = dinv[node];
  float v = xw[i] * di * di;       // self-loop: norm = dinv*1*dinv
  if (addBias) v += bias[f];
  out[i] = v;
}

// ---------- edge scatter: one wave per edge, FPL floats per lane (F = 32*FPL) ----------
template <int FPL>
__global__ void scatter_kernel(const float* __restrict__ xw, const int* __restrict__ src,
                               const int* __restrict__ dst, const float* __restrict__ ew,
                               const float* __restrict__ dinv, float* __restrict__ out, int E) {
  long long gid = (long long)blockIdx.x * blockDim.x + threadIdx.x;
  int e = (int)(gid >> 5);
  if (e >= E) return;
  int lane = (int)(gid & 31);
  int s = src[e];
  int d = dst[e];
  float nrm = dinv[s] * ew[e] * dinv[d];
  if (FPL == 2) {
    const float2 v = *(const float2*)(xw + (size_t)s * 64 + lane * 2);  // one b64 load
    float* op = out + (size_t)d * 64 + lane * 2;
    unsafeAtomicAdd(op,     v.x * nrm);
    unsafeAtomicAdd(op + 1, v.y * nrm);
  } else {
    const int F = 32 * FPL;
    const float* xp = xw + (size_t)s * F + lane * FPL;
    float*       op = out + (size_t)d * F + lane * FPL;
#pragma unroll
    for (int j = 0; j < FPL; ++j) unsafeAtomicAdd(op + j, xp[j] * nrm);
  }
}

// ---------- fused bias + ReLU ----------
__global__ void relu_bias_kernel(float* __restrict__ h, const float* __restrict__ b,
                                 int N, int F) {
  long long i = (long long)blockIdx.x * blockDim.x + threadIdx.x;
  if (i >= (long long)N * F) return;
  h[i] = fmaxf(h[i] + b[(int)(i % F)], 0.0f);
}

extern "C" void kernel_launch(void* const* d_in, const int* in_sizes, int n_in,
                              void* d_out, int out_size, void* d_ws, size_t ws_size,
                              hipStream_t stream) {
  const float* x  = (const float*)d_in[0];
  const int*   ei = (const int*)  d_in[1];
  const float* ew = (const float*)d_in[2];
  const float* W1 = (const float*)d_in[3];
  const float* b1 = (const float*)d_in[4];
  const float* W2 = (const float*)d_in[5];
  const float* b2 = (const float*)d_in[6];
  float* out = (float*)d_out;

  const int HID = in_sizes[4];              // 64
  const int CLS = in_sizes[6];              // 32
  const int FIN = in_sizes[3] / HID;        // 128
  const int N   = in_sizes[0] / FIN;        // 100000
  const int E   = in_sizes[2];              // 1600000
  const int* src = ei;
  const int* dst = ei + E;

  // workspace layout (floats, 64B-aligned chunks)
  float* ws = (float*)d_ws;
  size_t off = 0;
  float* dinv = ws + off; off += ((size_t)N + 15) & ~(size_t)15;
  float* xw   = ws + off; off += (((size_t)N * HID) + 15) & ~(size_t)15;
  float* h1   = ws + off;                   // N*HID accumulator, then h after relu
  float* hw   = xw;                         // reuse xw for layer-2 GEMM output (N*CLS <= N*HID)

  const int B = 256;

  // degree -> dinv
  init_deg_kernel<<<(N + B - 1) / B, B, 0, stream>>>(dinv, N);
  deg_accum_kernel<<<(E + B - 1) / B, B, 0, stream>>>(dst, ew, dinv, E);
  dinv_kernel<<<(N + B - 1) / B, B, 0, stream>>>(dinv, N);

  // layer 1: xw = x @ W1 (WMMA, NT=4 -> 64 output cols)
  {
    long long thr = (long long)(N / 16) * 32;
    gemm_bf16_wmma_kernel<4><<<(int)((thr + B - 1) / B), B, 0, stream>>>(x, W1, xw, N, FIN);
  }
  // self-loop init + edge scatter into h1
  {
    long long tot = (long long)N * HID;
    selfloop_init_kernel<64><<<(int)((tot + B - 1) / B), B, 0, stream>>>(xw, dinv, b1, h1, N, 0);
  }
  {
    long long thr = (long long)E * 32;
    scatter_kernel<2><<<(int)((thr + B - 1) / B), B, 0, stream>>>(xw, src, dst, ew, dinv, h1, E);
  }
  // h = relu(h1 + b1)
  {
    long long tot = (long long)N * HID;
    relu_bias_kernel<<<(int)((tot + B - 1) / B), B, 0, stream>>>(h1, b1, N, HID);
  }

  // layer 2: hw = h @ W2 (WMMA, NT=2 -> 32 output cols)
  {
    long long thr = (long long)(N / 16) * 32;
    gemm_bf16_wmma_kernel<2><<<(int)((thr + B - 1) / B), B, 0, stream>>>(h1, W2, hw, N, HID);
  }
  // self-loop init (+b2) directly into d_out, then edge scatter
  {
    long long tot = (long long)N * CLS;
    selfloop_init_kernel<32><<<(int)((tot + B - 1) / B), B, 0, stream>>>(hw, dinv, b2, out, N, 1);
  }
  {
    long long thr = (long long)E * 32;
    scatter_kernel<1><<<(int)((thr + B - 1) / B), B, 0, stream>>>(hw, src, dst, ew, dinv, out, E);
  }
}
